// OneToManyMultiheadMonotonicAttention_79834852098437
// MI455X (gfx1250) — compile-verified, matrix operand order checked by reading, not run on hardware
//
#include <hip/hip_runtime.h>

// ---------------------------------------------------------------------------
// MI455X (gfx1250) monotonic multihead attention, bf16 WMMA pipeline.
// ---------------------------------------------------------------------------

typedef __attribute__((ext_vector_type(16))) __bf16 v16bf;
typedef __attribute__((ext_vector_type(8)))  float  v8f;

constexpr int B_ = 2, TQ_ = 512, TK_ = 2048, E_ = 1024, H_ = 16, HD_ = 64;
constexpr float CLAMP_ = 5.0f;
constexpr float NEG_   = -1e4f;
constexpr float INV_NORM_ = 0.125f;   // 1/sqrt(64)
constexpr float LOG2E_ = 1.4426950408889634f;
constexpr float LN2_   = 0.6931471805599453f;

__device__ __forceinline__ v8f wmma_bf16(v16bf a, v16bf b, v8f c) {
  // (neg_a, A, neg_b, B, c_mod, C, reuse_a, reuse_b)
  return __builtin_amdgcn_wmma_f32_16x16x32_bf16(false, a, false, b, (short)0, c,
                                                 false, false);
}

// log(1+exp(x)) using raw v_exp_f32 (exp2) / v_log_f32 (log2).
// Safe here: x is either in [-5.01, 5.01] (clamped scores) or <= 0
// (logaddexp delta), so 1+2^(x*log2e) is in [1, ~150] — no denorm fixup
// or extended-precision ln2 correction needed on the serial path.
__device__ __forceinline__ float softplus_fast(float x) {
  return LN2_ * __builtin_amdgcn_logf(1.0f + __builtin_amdgcn_exp2f(x * LOG2E_));
}

__device__ __forceinline__ float logaddexpf_(float a, float b) {
  float m = fmaxf(a, b);
  return m + softplus_fast(fminf(a, b) - m);
}

__device__ __forceinline__ float redmax16(float v) {
#pragma unroll
  for (int off = 1; off < 16; off <<= 1) v = fmaxf(v, __shfl_xor(v, off, 32));
  return v;
}
__device__ __forceinline__ float redsum16(float v) {
#pragma unroll
  for (int off = 1; off < 16; off <<= 1) v += __shfl_xor(v, off, 32);
  return v;
}

// ---------------------------------------------------------------------------
// fp32 -> bf16 conversion (grid-stride)
// ---------------------------------------------------------------------------
__global__ void f32_to_bf16_kernel(const float* __restrict__ src,
                                   __bf16* __restrict__ dst, size_t n) {
  for (size_t i = blockIdx.x * (size_t)blockDim.x + threadIdx.x; i < n;
       i += (size_t)gridDim.x * blockDim.x)
    dst[i] = (__bf16)src[i];
}

// ---------------------------------------------------------------------------
// xvT[b][h][d][k] = xv[(b*TK+k)*E + h*HD + d]   (coalesced writes, L2 reads)
// ---------------------------------------------------------------------------
__global__ void transpose_v_kernel(const __bf16* __restrict__ xv,
                                   __bf16* __restrict__ xvT) {
  size_t n = (size_t)B_ * H_ * HD_ * TK_;
  for (size_t idx = blockIdx.x * (size_t)blockDim.x + threadIdx.x; idx < n;
       idx += (size_t)gridDim.x * blockDim.x) {
    int k  = (int)(idx & (TK_ - 1));
    size_t t = idx >> 11;              // / TK_
    int d  = (int)(t & (HD_ - 1));
    int bh = (int)(t >> 6);            // / HD_
    int h  = bh & (H_ - 1), b = bh >> 4;
    xvT[idx] = xv[((size_t)(b * TK_ + k)) * E_ + h * HD_ + d];
  }
}

// ---------------------------------------------------------------------------
// C[M,N] = A[M,K] * W[N,K]^T + bias.  Wave computes a 64x64 register tile:
// 16 v_wmma_f32_16x16x32_bf16 per 32-K step.  ld(A)=ld(W)=K.
// ---------------------------------------------------------------------------
template <bool OUTF32>
__global__ __launch_bounds__(256) void gemm_bias_kernel(
    const __bf16* __restrict__ A, const __bf16* __restrict__ W,
    const float* __restrict__ bias, void* __restrict__ Cout, int M, int N,
    int K) {
  int tilesN = N >> 6;
  int wave = blockIdx.x * 8 + (threadIdx.x >> 5);
  int total = (M >> 6) * tilesN;
  if (wave >= total) return;               // wave-uniform: EXEC stays all-ones
  int lane = threadIdx.x & 31;
  int tm = (wave / tilesN) << 6;
  int tn = (wave % tilesN) << 6;
  int row = lane & 15, hs = lane >> 4;

  v8f acc[4][4] = {};
  for (int k = 0; k < K; k += 32) {
    v16bf a[4], w[4];
#pragma unroll
    for (int i = 0; i < 4; ++i)
      a[i] = *(const v16bf*)(A + (size_t)(tm + 16 * i + row) * K + k + hs * 16);
#pragma unroll
    for (int j = 0; j < 4; ++j)
      w[j] = *(const v16bf*)(W + (size_t)(tn + 16 * j + row) * K + k + hs * 16);
#pragma unroll
    for (int i = 0; i < 4; ++i)
#pragma unroll
      for (int j = 0; j < 4; ++j) acc[i][j] = wmma_bf16(a[i], w[j], acc[i][j]);
  }
#pragma unroll
  for (int i = 0; i < 4; ++i)
#pragma unroll
    for (int j = 0; j < 4; ++j) {
      int col = tn + 16 * j + row;
      float bb = bias[col];
#pragma unroll
      for (int r = 0; r < 8; ++r) {
        int rr = tm + 16 * i + r + 8 * hs;
        float v = acc[i][j][r] + bb;
        if (OUTF32)
          ((float*)Cout)[(size_t)rr * N + col] = v;
        else
          ((__bf16*)Cout)[(size_t)rr * N + col] = (__bf16)v;
      }
    }
}

// ---------------------------------------------------------------------------
// Fused monotonic scan.  One 512-thread WG per (b,h) head; thread = query col.
// Per 16-key chunk: WMMA computes all 16x512 clamped scores -> log_sigmoid
// pair cached in LDS (fp16); then 16 serial logaddexp recurrence steps with
// the column shift passed through LDS.  phi rows staged in LDS (8 rows) and
// flushed coalesced into d_out's mono region (B,H,TQ,TK).
// ---------------------------------------------------------------------------
__global__ __launch_bounds__(512) void mono_scan_kernel(
    const __bf16* __restrict__ xqa, const __bf16* __restrict__ xk,
    float* __restrict__ mono) {
  __shared__ _Float16 lpS[16][TQ_];
  __shared__ _Float16 lqS[16][TQ_];
  __shared__ float phiB[TQ_][8];
  __shared__ float tbuf[TQ_];

  int bh = blockIdx.x;
  int b = bh >> 4, h = bh & (H_ - 1);
  const __bf16* Q  = xqa + ((size_t)b * TQ_) * E_ + h * HD_;
  const __bf16* Kp = xk + ((size_t)b * TK_) * E_ + h * HD_;
  float* out = mono + (size_t)bh * TQ_ * TK_;

  int tid = threadIdx.x;
  int lane = tid & 31, wv = tid >> 5;
  int row = lane & 15, hs = lane >> 4;

  float lphi = (tid == 0) ? 0.0f : NEG_;
  phiB[tid][0] = (tid == 0) ? 1.0f : 0.0f;   // phi row 0 = one-hot(col 0)

  for (int c = 0; c < TK_ / 16; ++c) {
    __syncthreads();  // previous chunk fully consumed; phiB[.][0] visible
    {
      // A = xk rows [16c,16c+16) x 64 dims ; B = xqa^T (dims x queries)
      v16bf a0 = *(const v16bf*)(Kp + (size_t)(16 * c + row) * E_ + hs * 16);
      v16bf a1 = *(const v16bf*)(Kp + (size_t)(16 * c + row) * E_ + 32 + hs * 16);
#pragma unroll
      for (int t = 0; t < 2; ++t) {
        int n0 = (wv * 2 + t) * 16;
        v16bf b0 = *(const v16bf*)(Q + (size_t)(n0 + row) * E_ + hs * 16);
        v16bf b1 = *(const v16bf*)(Q + (size_t)(n0 + row) * E_ + 32 + hs * 16);
        v8f cc = {};
        cc = wmma_bf16(a0, b0, cc);
        cc = wmma_bf16(a1, b1, cc);
#pragma unroll
        for (int r = 0; r < 8; ++r) {
          float s = cc[r];
          s = fminf(fmaxf(s, -CLAMP_), CLAMP_);
          int krow = r + 8 * hs, q = n0 + row;
          lpS[krow][q] = (_Float16)(-softplus_fast(-s));  // log_sigmoid(s)
          lqS[krow][q] = (_Float16)(-softplus_fast(s));   // log_sigmoid(-s)
        }
      }
    }
    __syncthreads();
    for (int r = 0; r < 16; ++r) {
      int i = 16 * c + r + 1;          // output row produced from score row i-1
      if (i >= TK_) break;             // uniform over the block
      float t = lphi + (float)lqS[r][tid];
      tbuf[tid] = t;
      __syncthreads();
      float tm = tid ? tbuf[tid - 1] : NEG_;
      float stay = lphi + (float)lpS[r][tid];
      lphi = logaddexpf_(stay, tm);
      int slot = i & 7;
      phiB[tid][slot] = __expf(lphi);
      __syncthreads();
      if (slot == 7) {                 // flush rows i-7..i, coalesced
        int kb = i - 7;
        for (int idx = tid; idx < TQ_ * 8; idx += 512) {
          int q = idx >> 3, kk = idx & 7;
          out[(size_t)q * TK_ + kb + kk] = phiB[q][kk];
        }
        __syncthreads();
      }
    }
  }
}

// ---------------------------------------------------------------------------
// Single-pass flash attention: P = softmax((mono + xq·xk^T)/8), O = P·xv.
// One wave per 16-query tile; 8 WMMA per 32-key step.  P is transposed from
// the C-layout into A-frag layout through a 1KB per-wave LDS stage.
// ---------------------------------------------------------------------------
__global__ __launch_bounds__(128) void flash_kernel(
    const __bf16* __restrict__ xq, const __bf16* __restrict__ xk,
    const __bf16* __restrict__ xvT, const float* __restrict__ mono,
    __bf16* __restrict__ xo) {
  __shared__ __bf16 pStage[4][16][32] __attribute__((aligned(32)));

  int wv = threadIdx.x >> 5, lane = threadIdx.x & 31;
  int wid = blockIdx.x * 4 + wv;       // 0..1023
  int qt = wid & 31;                   // TQ/16 = 32 tiles
  int bh = wid >> 5;
  int b = bh >> 4, h = bh & (H_ - 1);
  int row = lane & 15, hs = lane >> 4;

  const __bf16* Q  = xq + ((size_t)(b * TQ_ + qt * 16)) * E_ + h * HD_;
  const __bf16* Kp = xk + ((size_t)b * TK_) * E_ + h * HD_;
  const __bf16* V  = xvT + (size_t)bh * HD_ * TK_;
  const float* mon = mono + ((size_t)(bh * TQ_ + qt * 16)) * TK_;

  v16bf aq0 = *(const v16bf*)(Q + (size_t)row * E_ + hs * 16);
  v16bf aq1 = *(const v16bf*)(Q + (size_t)row * E_ + 32 + hs * 16);

  v8f acc[4] = {};
  float m8[8], l8[8];
#pragma unroll
  for (int r = 0; r < 8; ++r) { m8[r] = -3.0e38f; l8[r] = 0.0f; }
  __bf16 (*ps)[32] = pStage[wv];

  for (int kb = 0; kb < TK_; kb += 32) {
    float sv[2][8];
#pragma unroll
    for (int sub = 0; sub < 2; ++sub) {
      const __bf16* Kr = Kp + (size_t)(kb + sub * 16 + row) * E_;
      v16bf b0 = *(const v16bf*)(Kr + hs * 16);
      v16bf b1 = *(const v16bf*)(Kr + 32 + hs * 16);
      v8f cc = {};
      cc = wmma_bf16(aq0, b0, cc);
      cc = wmma_bf16(aq1, b1, cc);
#pragma unroll
      for (int r = 0; r < 8; ++r) {
        int qr = r + 8 * hs;
        sv[sub][r] =
            (cc[r] + mon[(size_t)qr * TK_ + kb + sub * 16 + row]) * INV_NORM_;
      }
    }
#pragma unroll
    for (int r = 0; r < 8; ++r) {
      float tmax = redmax16(fmaxf(sv[0][r], sv[1][r]));
      float mn = fmaxf(m8[r], tmax);
      float sc = __builtin_amdgcn_exp2f((m8[r] - mn) * LOG2E_);
      m8[r] = mn;
      l8[r] *= sc;
#pragma unroll
      for (int j = 0; j < 4; ++j) acc[j][r] *= sc;
      float p0 = __builtin_amdgcn_exp2f((sv[0][r] - mn) * LOG2E_);
      float p1 = __builtin_amdgcn_exp2f((sv[1][r] - mn) * LOG2E_);
      ps[r + 8 * hs][row] = (__bf16)p0;
      ps[r + 8 * hs][16 + row] = (__bf16)p1;
      l8[r] += redsum16(p0 + p1);
    }
    // intra-wave LDS RAW: stores above feed the A-frag load below
    asm volatile("s_wait_dscnt 0x0" ::: "memory");
    v16bf pA = *(const v16bf*)&ps[row][hs * 16];
#pragma unroll
    for (int j = 0; j < 4; ++j) {
      v16bf vf =
          *(const v16bf*)(V + (size_t)(j * 16 + row) * TK_ + kb + hs * 16);
      acc[j] = wmma_bf16(pA, vf, acc[j]);
    }
  }
#pragma unroll
  for (int r = 0; r < 8; ++r) l8[r] = 1.0f / l8[r];
  __bf16* O = xo + ((size_t)(b * TQ_ + qt * 16)) * E_ + h * HD_;
#pragma unroll
  for (int j = 0; j < 4; ++j)
#pragma unroll
    for (int r = 0; r < 8; ++r)
      O[(size_t)(r + 8 * hs) * E_ + j * 16 + row] = (__bf16)(acc[j][r] * l8[r]);
}

// ---------------------------------------------------------------------------
extern "C" void kernel_launch(void* const* d_in, const int* in_sizes, int n_in,
                              void* d_out, int out_size, void* d_ws,
                              size_t ws_size, hipStream_t stream) {
  (void)in_sizes; (void)n_in; (void)out_size; (void)ws_size;
  const float* q_f  = (const float*)d_in[0];
  const float* k_f  = (const float*)d_in[1];
  const float* v_f  = (const float*)d_in[2];
  const float* wi_f = (const float*)d_in[3];
  const float* bi_f = (const float*)d_in[4];
  const float* wo_f = (const float*)d_in[5];
  const float* bo_f = (const float*)d_in[6];

  char* ws = (char*)d_ws;
  size_t off = 0;
  auto carve = [&](size_t elems) -> __bf16* {
    __bf16* p = (__bf16*)(ws + off);
    off += ((elems * 2 + 255) / 256) * 256;
    return p;
  };
  __bf16* q_bf    = carve((size_t)B_ * TQ_ * E_);
  __bf16* k_bf    = carve((size_t)B_ * TK_ * E_);
  __bf16* v_bf    = carve((size_t)B_ * TK_ * E_);
  __bf16* wi_bf   = carve((size_t)4 * E_ * E_);
  __bf16* wo_bf   = carve((size_t)E_ * E_);
  __bf16* xqa_bf  = carve((size_t)B_ * TQ_ * E_);
  __bf16* xq_bf   = carve((size_t)B_ * TQ_ * E_);
  __bf16* xk_bf   = carve((size_t)B_ * TK_ * E_);
  __bf16* xv_bf   = carve((size_t)B_ * TK_ * E_);
  __bf16* xvT_bf  = carve((size_t)B_ * H_ * HD_ * TK_);
  __bf16* xo_bf   = carve((size_t)B_ * TQ_ * E_);

  float* out_xo = (float*)d_out;
  float* mono   = out_xo + (size_t)B_ * TQ_ * E_;   // (B,1,H,TQ,TK)

  auto cvt = [&](const float* src, __bf16* dst, size_t n) {
    int blocks = (int)((n + 2047) / 2048);
    f32_to_bf16_kernel<<<blocks, 256, 0, stream>>>(src, dst, n);
  };
  cvt(q_f, q_bf, (size_t)B_ * TQ_ * E_);
  cvt(k_f, k_bf, (size_t)B_ * TK_ * E_);
  cvt(v_f, v_bf, (size_t)B_ * TK_ * E_);
  cvt(wi_f, wi_bf, (size_t)4 * E_ * E_);
  cvt(wo_f, wo_bf, (size_t)E_ * E_);

  auto gemm = [&](const __bf16* A, const __bf16* W, const float* bias, void* C,
                  int M, int N, int K, bool f32out) {
    int tiles = (M / 64) * (N / 64);
    int blocks = (tiles + 7) / 8;
    if (f32out)
      gemm_bias_kernel<true><<<blocks, 256, 0, stream>>>(A, W, bias, C, M, N, K);
    else
      gemm_bias_kernel<false><<<blocks, 256, 0, stream>>>(A, W, bias, C, M, N, K);
  };
  // in_proj rows: [qwa | qw | kw | vw]
  gemm(q_bf, wi_bf,                      bi_f,          xqa_bf, B_ * TQ_, E_, E_, false);
  gemm(q_bf, wi_bf + (size_t)E_ * E_,    bi_f + E_,     xq_bf,  B_ * TQ_, E_, E_, false);
  gemm(k_bf, wi_bf + (size_t)2 * E_ * E_, bi_f + 2 * E_, xk_bf, B_ * TK_, E_, E_, false);
  gemm(v_bf, wi_bf + (size_t)3 * E_ * E_, bi_f + 3 * E_, xv_bf, B_ * TK_, E_, E_, false);

  {
    size_t n = (size_t)B_ * H_ * HD_ * TK_;
    transpose_v_kernel<<<(int)((n + 2047) / 2048), 256, 0, stream>>>(xv_bf, xvT_bf);
  }

  mono_scan_kernel<<<B_ * H_, 512, 0, stream>>>(xqa_bf, xk_bf, mono);

  flash_kernel<<<(B_ * H_ * (TQ_ / 16)) / 4, 128, 0, stream>>>(xq_bf, xk_bf,
                                                               xvT_bf, mono,
                                                               xo_bf);

  gemm(xo_bf, wo_bf, bo_f, out_xo, B_ * TQ_, E_, E_, true);
}
